// MoELayer_51977694216450
// MI455X (gfx1250) — matrix operand order repaired and checked by reading, not compiled, not run on hardware
//
#include <hip/hip_runtime.h>
#include <hip/hip_bf16.h>

// ---------------------------------------------------------------------------
// MoE SwiGLU FFN for MI455X (gfx1250, wave32, WMMA bf16 path)
//   S=8192 tokens, C=1024, E=8 experts, top-2, HID=4096
// Pipeline: gate -> plan (128-aligned expert buckets) -> scatter/gather ->
//           GEMM1 (x@W1, x@W3 fused SwiGLU) -> GEMM2 (h@W2 * gate weight) ->
//           combine (sum the 2 expert rows per token).
// ---------------------------------------------------------------------------

typedef __attribute__((ext_vector_type(16))) __bf16 bf16x16;
typedef __attribute__((ext_vector_type(8)))  __bf16 bf16x8;
typedef __attribute__((ext_vector_type(8)))  float  f32x8;

constexpr int S_       = 8192;   // B*T tokens
constexpr int C_       = 1024;   // model dim
constexpr int E_       = 8;      // experts
constexpr int H_       = 4096;   // hidden dim
constexpr int K_       = 2;      // top-k
constexpr int PADROWS_ = 17408;  // 16384 assignments + 8*128 padding
constexpr int MAXTILES_= 136;    // PADROWS_/128

__device__ __forceinline__ f32x8 wmma_bf16(bf16x16 a, bf16x16 b, f32x8 c) {
  // v_wmma_f32_16x16x32_bf16: D = A(16x32) x B(32x16) + C(16x16 f32)
  return __builtin_amdgcn_wmma_f32_16x16x32_bf16(
      /*neg_a=*/false, a, /*neg_b=*/false, b,
      /*c_mod=*/(short)0, c, /*reuse_a=*/false, /*reuse_b=*/false);
}

// A fragment: 16x32 bf16 tile from LDS row-major [rows][32].
// ISA layout: lanes 0-15 hold row M=lane, K=0..7 & 16..23; lanes 16-31 hold
// row M=lane-16, K=8..15 & 24..31  -> two b128 LDS reads per lane.
__device__ __forceinline__ bf16x16 frag_a(const __bf16* lds, int m0, int lane) {
  const int row  = m0 + (lane & 15);
  const int koff = (lane >> 4) * 8;
  union { bf16x16 v; bf16x8 h[2]; } u;
  u.h[0] = *(const bf16x8*)(lds + row * 32 + koff);
  u.h[1] = *(const bf16x8*)(lds + row * 32 + koff + 16);
  return u.v;
}

// B fragment: 32x16 bf16 tile, staged *transposed* in LDS as [N][32]; same K
// order as A: lanes 0-15 -> K=0..7 & 16..23 of col N=lane, lanes 16-31 ->
// K=8..15 & 24..31 of col N=lane-16.
__device__ __forceinline__ bf16x16 frag_b(const __bf16* lds, int n0, int lane) {
  const int col  = n0 + (lane & 15);
  const int koff = (lane >> 4) * 8;
  union { bf16x16 v; bf16x8 h[2]; } u;
  u.h[0] = *(const bf16x8*)(lds + col * 32 + koff);
  u.h[1] = *(const bf16x8*)(lds + col * 32 + koff + 16);
  return u.v;
}

// ---------------------------------------------------------------------------
// Weight convert + transpose: in (E,R,Q) f32 -> out (E,Q,R) bf16.
// Makes every GEMM consume B^T with K contiguous (uint4-coalesced staging).
// ---------------------------------------------------------------------------
__global__ void transpose_convert(const float* __restrict__ in,
                                  __bf16* __restrict__ out, int R, int Q) {
  __shared__ float tile[32][33];
  const size_t base = (size_t)blockIdx.z * (size_t)R * (size_t)Q;
  const int q0 = blockIdx.x * 32, r0 = blockIdx.y * 32;
  const int tx = threadIdx.x, ty = threadIdx.y;
#pragma unroll
  for (int i = 0; i < 32; i += 8)
    tile[ty + i][tx] = in[base + (size_t)(r0 + ty + i) * Q + (q0 + tx)];
  __syncthreads();
#pragma unroll
  for (int i = 0; i < 32; i += 8)
    out[base + (size_t)(q0 + ty + i) * R + (r0 + tx)] = (__bf16)tile[tx][ty + i];
}

// ---------------------------------------------------------------------------
// Gate: one wave per token. logits = x @ wg_w + wg_b, softmax, top-2.
// ---------------------------------------------------------------------------
__global__ void gate_kernel(const float* __restrict__ x,
                            const float* __restrict__ wgw,
                            const float* __restrict__ wgb,
                            float* __restrict__ probs,
                            int* __restrict__ idx_out,
                            float* __restrict__ val_out,
                            int* __restrict__ cnt,
                            int* __restrict__ load_cnt) {
  const int lane = threadIdx.x & 31;
  const int t = blockIdx.x * 8 + (threadIdx.x >> 5);
  float acc[E_];
#pragma unroll
  for (int e = 0; e < E_; ++e) acc[e] = 0.f;
  for (int c = lane; c < C_; c += 32) {
    const float xv = x[(size_t)t * C_ + c];
    const float4 wa = *(const float4*)(wgw + (size_t)c * E_);
    const float4 wb = *(const float4*)(wgw + (size_t)c * E_ + 4);
    acc[0] += xv * wa.x; acc[1] += xv * wa.y; acc[2] += xv * wa.z; acc[3] += xv * wa.w;
    acc[4] += xv * wb.x; acc[5] += xv * wb.y; acc[6] += xv * wb.z; acc[7] += xv * wb.w;
  }
#pragma unroll
  for (int e = 0; e < E_; ++e) {
#pragma unroll
    for (int off = 16; off > 0; off >>= 1) acc[e] += __shfl_xor(acc[e], off, 32);
  }
  if (lane == 0) {
    float p[E_];
    float m = -1e30f;
#pragma unroll
    for (int e = 0; e < E_; ++e) { p[e] = acc[e] + wgb[e]; m = fmaxf(m, p[e]); }
    float ssum = 0.f;
#pragma unroll
    for (int e = 0; e < E_; ++e) { p[e] = __expf(p[e] - m); ssum += p[e]; }
    const float inv = 1.0f / ssum;
#pragma unroll
    for (int e = 0; e < E_; ++e) { p[e] *= inv; probs[(size_t)t * E_ + e] = p[e]; }
    int i0 = 0;
#pragma unroll
    for (int e = 1; e < E_; ++e) if (p[e] > p[i0]) i0 = e;        // first-max tie rule
    int i1 = (i0 == 0) ? 1 : 0;
#pragma unroll
    for (int e = 0; e < E_; ++e) if (e != i0 && p[e] > p[i1]) i1 = e;
    idx_out[2 * t]     = i0;  val_out[2 * t]     = p[i0];
    idx_out[2 * t + 1] = i1;  val_out[2 * t + 1] = p[i1];
    atomicAdd(&load_cnt[i0], 1);
    atomicAdd(&cnt[i0], 1);
    atomicAdd(&cnt[i1], 1);
  }
}

// Deterministic column sums of probs -> importance (one block per expert).
__global__ void importance_kernel(const float* __restrict__ probs,
                                  float* __restrict__ imp) {
  __shared__ float red[256];
  const int e = blockIdx.x;
  float s = 0.f;
  for (int t = threadIdx.x; t < S_; t += 256) s += probs[(size_t)t * E_ + e];
  red[threadIdx.x] = s;
  __syncthreads();
  for (int w = 128; w > 0; w >>= 1) {
    if ((int)threadIdx.x < w) red[threadIdx.x] += red[threadIdx.x + w];
    __syncthreads();
  }
  if (threadIdx.x == 0) imp[e] = red[0];
}

// Build 128-row-aligned expert buckets + static tile table; emit aux scalar.
__global__ void plan_kernel(const int* __restrict__ cnt,
                            const int* __restrict__ load_cnt,
                            const float* __restrict__ imp,
                            int* __restrict__ offs, int* __restrict__ cursor,
                            int4* __restrict__ table, float* __restrict__ aux_out) {
  if (threadIdx.x != 0 || blockIdx.x != 0) return;
  int tile = 0, off = 0;
  for (int e = 0; e < E_; ++e) {
    offs[e] = off;
    cursor[e] = 0;
    const int n = cnt[e];
    const int ntile = (n + 127) >> 7;
    for (int i = 0; i < ntile; ++i) {
      int valid = n - i * 128;
      if (valid > 128) valid = 128;
      table[tile++] = make_int4(e, off + i * 128, valid, 1);
    }
    off += ntile * 128;
  }
  for (; tile < MAXTILES_; ++tile) table[tile] = make_int4(0, 0, 0, 0);
  const float invS = 1.0f / (float)S_;
  float aux = 0.f;
  for (int e = 0; e < E_; ++e)
    aux += (imp[e] * invS) * ((float)load_cnt[e] * invS);
  *aux_out = aux * (float)E_;
}

__global__ void scatter_kernel(const int* __restrict__ idx_out,
                               const float* __restrict__ val_out,
                               const int* __restrict__ offs,
                               int* __restrict__ cursor,
                               int* __restrict__ slot_of,
                               float* __restrict__ slot_wt) {
  const int i = blockIdx.x * blockDim.x + threadIdx.x;
  if (i >= S_ * K_) return;
  const int e = idx_out[i];
  const int pos = offs[e] + atomicAdd(&cursor[e], 1);
  slot_of[i] = pos;
  slot_wt[pos] = val_out[i];
}

// Gather token rows into expert buckets, converting f32 -> bf16 on the fly.
__global__ void gather_kernel(const float* __restrict__ x,
                              const int* __restrict__ slot_of,
                              __bf16* __restrict__ Xg) {
  const int lane = threadIdx.x & 31;
  const int i = blockIdx.x * 8 + (threadIdx.x >> 5);   // assignment id
  const int t = i >> 1;
  const int pos = slot_of[i];
  for (int c = lane; c < C_; c += 32)
    Xg[(size_t)pos * C_ + c] = (__bf16)x[(size_t)t * C_ + c];
}

// ---------------------------------------------------------------------------
// GEMM1: H = silu(Xg @ W1^T + b1) * (Xg @ W3^T + b3), bf16 in / f32 acc.
// Block tile 128(M) x 64(N), 8 waves of 32x32, double-buffered LDS.
// ---------------------------------------------------------------------------
__global__ __launch_bounds__(256) void gemm1_swiglu(
    const __bf16* __restrict__ Xg, const __bf16* __restrict__ W1t,
    const __bf16* __restrict__ W3t, const float* __restrict__ b1,
    const float* __restrict__ b3, __bf16* __restrict__ Hg,
    const int4* __restrict__ table) {
  __shared__ __bf16 sA[2][128 * 32];
  __shared__ __bf16 sB1[2][64 * 32];
  __shared__ __bf16 sB3[2][64 * 32];

  const int4 te = table[blockIdx.x];
  if (te.w == 0) return;
  const int e = te.x, row0 = te.y, nvalid = te.z;
  const int tid = threadIdx.x, lane = tid & 31, wave = tid >> 5;
  const int wm = (wave >> 1) * 32;           // 0,32,64,96
  const int wn = (wave & 1) * 32;            // 0,32
  const int n0g = blockIdx.y * 64;

  const __bf16* gA  = Xg  + (size_t)row0 * C_;
  const __bf16* gB1 = W1t + ((size_t)e * H_ + n0g) * C_;
  const __bf16* gB3 = W3t + ((size_t)e * H_ + n0g) * C_;

  const int ar = tid >> 1, ah = (tid & 1) * 16;                // A: 2 thr/row
  const int br = (tid & 127) >> 1, bh = ((tid & 127) & 1) * 16; // B: 2 thr/row
  const __bf16* gB = (tid < 128) ? gB1 : gB3;

  // Prime buffer 0.
  *(uint4*)(&sA[0][ar * 32 + ah]) = *(const uint4*)(gA + (size_t)ar * C_ + ah);
  {
    __bf16* sB = (tid < 128) ? &sB1[0][0] : &sB3[0][0];
    *(uint4*)(&sB[br * 32 + bh]) = *(const uint4*)(gB + (size_t)br * C_ + bh);
  }
  __syncthreads();

  f32x8 acc1[2][2] = {};
  f32x8 acc3[2][2] = {};

  const int KT = C_ / 32;
  for (int kk = 0; kk < KT; ++kk) {
    const int cur = kk & 1;
    if (kk + 1 < KT) {
      const int nxt = cur ^ 1;
      const int k0n = (kk + 1) * 32;
      *(uint4*)(&sA[nxt][ar * 32 + ah]) =
          *(const uint4*)(gA + (size_t)ar * C_ + k0n + ah);
      __bf16* sB = (tid < 128) ? &sB1[nxt][0] : &sB3[nxt][0];
      *(uint4*)(&sB[br * 32 + bh]) =
          *(const uint4*)(gB + (size_t)br * C_ + k0n + bh);
      if (kk + 2 < KT)                       // global_prefetch_b8 one tile ahead
        __builtin_prefetch(gA + (size_t)ar * C_ + k0n + 32 + ah, 0, 0);
    }
    bf16x16 af[2], bf1[2], bf3[2];
#pragma unroll
    for (int i = 0; i < 2; ++i) af[i] = frag_a(&sA[cur][0], wm + i * 16, lane);
#pragma unroll
    for (int j = 0; j < 2; ++j) {
      bf1[j] = frag_b(&sB1[cur][0], wn + j * 16, lane);
      bf3[j] = frag_b(&sB3[cur][0], wn + j * 16, lane);
    }
#pragma unroll
    for (int i = 0; i < 2; ++i)
#pragma unroll
      for (int j = 0; j < 2; ++j) {
        acc1[i][j] = wmma_bf16(af[i], bf1[j], acc1[i][j]);
        acc3[i][j] = wmma_bf16(af[i], bf3[j], acc3[i][j]);
      }
    __syncthreads();
  }

  // Fused SwiGLU epilogue. C/D layout: VGPR r -> M = r + (lane>=16 ? 8 : 0),
  // N = lane & 15.
  const int mbase = wm + ((lane >> 4) * 8);
  const int ncol  = lane & 15;
#pragma unroll
  for (int i = 0; i < 2; ++i)
#pragma unroll
    for (int j = 0; j < 2; ++j) {
      const int ng = n0g + wn + j * 16 + ncol;
      const float bb1 = b1[e * H_ + ng];
      const float bb3 = b3[e * H_ + ng];
#pragma unroll
      for (int r = 0; r < 8; ++r) {
        const int m = mbase + i * 16 + r;
        if (m < nvalid) {
          const float v1 = acc1[i][j][r] + bb1;
          const float v3 = acc3[i][j][r] + bb3;
          const float h = (v1 / (1.0f + __expf(-v1))) * v3;  // silu(v1)*v3
          Hg[(size_t)(row0 + m) * H_ + ng] = (__bf16)h;
        }
      }
    }
}

// ---------------------------------------------------------------------------
// GEMM2: Yg = (Hg @ W2^T + b2) * gate_weight(row).
// ---------------------------------------------------------------------------
__global__ __launch_bounds__(256) void gemm2_scale(
    const __bf16* __restrict__ Hg, const __bf16* __restrict__ W2t,
    const float* __restrict__ b2, const float* __restrict__ slot_wt,
    float* __restrict__ Yg, const int4* __restrict__ table) {
  __shared__ __bf16 sA[2][128 * 32];
  __shared__ __bf16 sB[2][64 * 32];

  const int4 te = table[blockIdx.x];
  if (te.w == 0) return;
  const int e = te.x, row0 = te.y, nvalid = te.z;
  const int tid = threadIdx.x, lane = tid & 31, wave = tid >> 5;
  const int wm = (wave >> 1) * 32;
  const int wn = (wave & 1) * 32;
  const int n0g = blockIdx.y * 64;

  const __bf16* gA = Hg  + (size_t)row0 * H_;
  const __bf16* gB = W2t + ((size_t)e * C_ + n0g) * H_;

  const int ar = tid >> 1, ah = (tid & 1) * 16;
  const int br = (tid & 127) >> 1, bh = ((tid & 127) & 1) * 16;

  *(uint4*)(&sA[0][ar * 32 + ah]) = *(const uint4*)(gA + (size_t)ar * H_ + ah);
  if (tid < 128)
    *(uint4*)(&sB[0][br * 32 + bh]) = *(const uint4*)(gB + (size_t)br * H_ + bh);
  __syncthreads();

  f32x8 acc[2][2] = {};

  const int KT = H_ / 32;
  for (int kk = 0; kk < KT; ++kk) {
    const int cur = kk & 1;
    if (kk + 1 < KT) {
      const int nxt = cur ^ 1;
      const int k0n = (kk + 1) * 32;
      *(uint4*)(&sA[nxt][ar * 32 + ah]) =
          *(const uint4*)(gA + (size_t)ar * H_ + k0n + ah);
      if (tid < 128)
        *(uint4*)(&sB[nxt][br * 32 + bh]) =
            *(const uint4*)(gB + (size_t)br * H_ + k0n + bh);
      if (kk + 2 < KT)
        __builtin_prefetch(gA + (size_t)ar * H_ + k0n + 32 + ah, 0, 0);
    }
    bf16x16 af[2], bfm[2];
#pragma unroll
    for (int i = 0; i < 2; ++i) af[i] = frag_a(&sA[cur][0], wm + i * 16, lane);
#pragma unroll
    for (int j = 0; j < 2; ++j) bfm[j] = frag_b(&sB[cur][0], wn + j * 16, lane);
#pragma unroll
    for (int i = 0; i < 2; ++i)
#pragma unroll
      for (int j = 0; j < 2; ++j)
        acc[i][j] = wmma_bf16(af[i], bfm[j], acc[i][j]);
    __syncthreads();
  }

  const int mbase = wm + ((lane >> 4) * 8);
  const int ncol  = lane & 15;
#pragma unroll
  for (int i = 0; i < 2; ++i)
#pragma unroll
    for (int j = 0; j < 2; ++j) {
      const int ng = n0g + wn + j * 16 + ncol;
      const float bias = b2[e * C_ + ng];
#pragma unroll
      for (int r = 0; r < 8; ++r) {
        const int m = mbase + i * 16 + r;
        if (m < nvalid) {
          const float wgt = slot_wt[row0 + m];
          Yg[(size_t)(row0 + m) * C_ + ng] = (acc[i][j][r] + bias) * wgt;
        }
      }
    }
}

// y[t] = Yg[slot(t,0)] + Yg[slot(t,1)]  (weights already applied; no atomics)
__global__ void combine_kernel(const float* __restrict__ Yg,
                               const int* __restrict__ slot_of,
                               float* __restrict__ y) {
  const int t = blockIdx.x;
  const int p0 = slot_of[2 * t], p1 = slot_of[2 * t + 1];
  const float4* a = (const float4*)(Yg + (size_t)p0 * C_);
  const float4* b = (const float4*)(Yg + (size_t)p1 * C_);
  float4* o = (float4*)(y + (size_t)t * C_);
  const int c = threadIdx.x;                 // C_/4 == 256
  const float4 va = a[c], vb = b[c];
  o[c] = make_float4(va.x + vb.x, va.y + vb.y, va.z + vb.z, va.w + vb.w);
}

__global__ void init_kernel(int* cnt, int* load_cnt) {
  if (threadIdx.x < E_) { cnt[threadIdx.x] = 0; load_cnt[threadIdx.x] = 0; }
}

// ---------------------------------------------------------------------------
extern "C" void kernel_launch(void* const* d_in, const int* in_sizes, int n_in,
                              void* d_out, int out_size, void* d_ws, size_t ws_size,
                              hipStream_t stream) {
  (void)in_sizes; (void)n_in; (void)out_size; (void)ws_size;
  const float* x   = (const float*)d_in[0];
  const float* wgw = (const float*)d_in[1];
  const float* wgb = (const float*)d_in[2];
  const float* w1  = (const float*)d_in[3];
  const float* b1  = (const float*)d_in[4];
  const float* w3  = (const float*)d_in[5];
  const float* b3  = (const float*)d_in[6];
  const float* w2  = (const float*)d_in[7];
  const float* b2  = (const float*)d_in[8];
  float* y = (float*)d_out;                  // S_*C_ outputs + aux scalar

  char* p = (char*)d_ws;
  auto take = [&](size_t bytes) -> char* {
    char* r = p;
    p += (bytes + 255) & ~(size_t)255;
    return r;
  };
  __bf16* W1t = (__bf16*)take((size_t)E_ * H_ * C_ * 2);   // (E,HID,C) bf16
  __bf16* W3t = (__bf16*)take((size_t)E_ * H_ * C_ * 2);   // (E,HID,C) bf16
  __bf16* W2t = (__bf16*)take((size_t)E_ * C_ * H_ * 2);   // (E,C,HID) bf16
  __bf16* Xg  = (__bf16*)take((size_t)PADROWS_ * C_ * 2);  // gathered tokens
  __bf16* Hg  = (__bf16*)take((size_t)PADROWS_ * H_ * 2);  // SwiGLU hidden
  float*  Yg  = (float*) take((size_t)PADROWS_ * C_ * 4);  // weighted outputs
  float*  probs   = (float*)take((size_t)S_ * E_ * 4);
  int*    idx_out = (int*)  take((size_t)S_ * K_ * 4);
  float*  val_out = (float*)take((size_t)S_ * K_ * 4);
  int*    slot_of = (int*)  take((size_t)S_ * K_ * 4);
  float*  slot_wt = (float*)take((size_t)PADROWS_ * 4);
  int*    cnt     = (int*)  take(256);
  int*    load_c  = (int*)  take(256);
  int*    cursor  = (int*)  take(256);
  int*    offs    = (int*)  take(256);
  float*  imp     = (float*)take(256);
  int4*   table   = (int4*) take((size_t)MAXTILES_ * sizeof(int4));

  init_kernel<<<1, 32, 0, stream>>>(cnt, load_c);
  transpose_convert<<<dim3(H_/32, C_/32, E_), dim3(32, 8), 0, stream>>>(w1, W1t, C_, H_);
  transpose_convert<<<dim3(H_/32, C_/32, E_), dim3(32, 8), 0, stream>>>(w3, W3t, C_, H_);
  transpose_convert<<<dim3(C_/32, H_/32, E_), dim3(32, 8), 0, stream>>>(w2, W2t, H_, C_);
  gate_kernel<<<S_/8, 256, 0, stream>>>(x, wgw, wgb, probs, idx_out, val_out, cnt, load_c);
  importance_kernel<<<E_, 256, 0, stream>>>(probs, imp);
  plan_kernel<<<1, 1, 0, stream>>>(cnt, load_c, imp, offs, cursor, table,
                                   y + (size_t)S_ * C_);
  scatter_kernel<<<(S_ * K_) / 256, 256, 0, stream>>>(idx_out, val_out, offs, cursor,
                                                      slot_of, slot_wt);
  gather_kernel<<<(S_ * K_) / 8, 256, 0, stream>>>(x, slot_of, Xg);
  gemm1_swiglu<<<dim3(MAXTILES_, H_/64), 256, 0, stream>>>(Xg, W1t, W3t, b1, b3, Hg, table);
  gemm2_scale<<<dim3(MAXTILES_, C_/64), 256, 0, stream>>>(Hg, W2t, b2, slot_wt, Yg, table);
  combine_kernel<<<S_, 256, 0, stream>>>(Yg, slot_of, y);
}
